// CausalSelfAttention_6708738917040
// MI455X (gfx1250) — matrix operand order repaired and checked by reading, not compiled, not run on hardware
//
#include <hip/hip_runtime.h>

#define DIM   1024
#define SEQ   2048
#define BATCH 2
#define NH    16
#define HD    64
#define NROWS (BATCH*SEQ)   // 4096
#define NQKV  (3*DIM)       // 3072

typedef __bf16 bf16_t;
typedef __bf16 v16bf __attribute__((ext_vector_type(16)));
typedef __bf16 v8bf  __attribute__((ext_vector_type(8)));
typedef float  v8f   __attribute__((ext_vector_type(8)));

union frag_u { v16bf v; v8bf h[2]; };

// Build a 16-element bf16 WMMA operand from two 16-byte runs.
__device__ __forceinline__ v16bf make_frag(const bf16_t* p0, const bf16_t* p1) {
  frag_u u;
  u.h[0] = *(const v8bf*)p0;
  u.h[1] = *(const v8bf*)p1;
  return u.v;
}

// LDS byte offset of a __shared__ object (addrspace(3) pointer value).
__device__ __forceinline__ unsigned lds_addr(const void* p) {
  return (unsigned)(unsigned long long)
      (const __attribute__((address_space(3))) char*)p;
}

// CDNA5 async global->LDS 16-byte copy (per lane), tracked by ASYNCcnt.
__device__ __forceinline__ void async_ld16(unsigned loff, const void* gp) {
  asm volatile("global_load_async_to_lds_b128 %0, %1, off"
               :: "v"(loff), "v"((unsigned long long)gp) : "memory");
}
__device__ __forceinline__ void wait_async0() {
  asm volatile("s_wait_asynccnt 0" ::: "memory");
}

// ---------------------------------------------------------------------------
// Prep 1: x f32 -> bf16, straight layout. 8 elements/thread, vectorized.
// ---------------------------------------------------------------------------
__global__ __launch_bounds__(256)
void cvt_x_kernel(const float* __restrict__ x, bf16_t* __restrict__ xb)
{
  size_t i = (size_t)blockIdx.x*256 + threadIdx.x;   // uint4 (8 bf16) index
  const float4* src = (const float4*)x;
  float4 a = src[2*i], b = src[2*i + 1];
  union { uint4 u; bf16_t h[8]; } o;
  o.h[0] = (bf16_t)a.x; o.h[1] = (bf16_t)a.y;
  o.h[2] = (bf16_t)a.z; o.h[3] = (bf16_t)a.w;
  o.h[4] = (bf16_t)b.x; o.h[5] = (bf16_t)b.y;
  o.h[6] = (bf16_t)b.z; o.h[7] = (bf16_t)b.w;
  ((uint4*)xb)[i] = o.u;
}

// ---------------------------------------------------------------------------
// Prep 2: W [K][N] f32 -> Wt [N][K] bf16 (transpose via LDS 64x64 tile).
// ---------------------------------------------------------------------------
__global__ __launch_bounds__(256)
void cvt_wt_kernel(const float* __restrict__ W, bf16_t* __restrict__ Wt,
                   int K, int N)
{
  __shared__ __align__(16) bf16_t Ts[64*72];         // [n][k], padded pitch
  const int tid = threadIdx.x;
  const int kTile = blockIdx.y*64, nTile = blockIdx.x*64;
  #pragma unroll
  for (int i = 0; i < 4; ++i) {                      // 1024 float4 per tile
    int e  = tid + i*256;
    int kl = e >> 4, n4 = (e & 15)*4;
    float4 w = *(const float4*)&W[(size_t)(kTile + kl)*N + nTile + n4];
    Ts[(n4+0)*72 + kl] = (bf16_t)w.x;
    Ts[(n4+1)*72 + kl] = (bf16_t)w.y;
    Ts[(n4+2)*72 + kl] = (bf16_t)w.z;
    Ts[(n4+3)*72 + kl] = (bf16_t)w.w;
  }
  __syncthreads();
  #pragma unroll
  for (int i = 0; i < 2; ++i) {                      // 512 uint4 out per tile
    int e = tid + i*256;
    int n = e >> 3, seg = (e & 7)*8;
    *(uint4*)&Wt[(size_t)(nTile + n)*K + kTile + seg] =
        *(const uint4*)&Ts[n*72 + seg];
  }
}

// ---------------------------------------------------------------------------
// Shared GEMM core: 64x64 tile/WG, 4 waves, 16 rows/wave, K-step 64.
// B tile ([n][k]) double-buffered in LDS via ASYNC global->LDS copies issued
// before the WMMA burst; one s_wait_asynccnt + one barrier per K-step.
// A-fragments load direct from global (wave-private rows).
// ---------------------------------------------------------------------------
#define GEMM_CORE(Wmat)                                                        \
  const int tid  = threadIdx.x;                                                \
  const int wave = tid >> 5, lane = tid & 31;                                  \
  const int lmod = lane & 15, g = lane >> 4;                                   \
  const int mTile = blockIdx.y * 64;                                           \
  const int nTile = blockIdx.x * 64;                                           \
  v8f acc[4];                                                                  \
  _Pragma("unroll") for (int nb = 0; nb < 4; ++nb) acc[nb] = {};               \
  const unsigned wbase = lds_addr(Wst);                                        \
  auto asyncB = [&](int kk, int buf) {                                         \
    _Pragma("unroll") for (int i = 0; i < 4; ++i) {                            \
      int e = tid + i*128; int n = e >> 3, seg = (e & 7)*8;                    \
      async_ld16(wbase + (unsigned)(buf*4096 + n*64 + seg)*2u,                 \
                 &Wmat[(size_t)(nTile + n)*DIM + kk + seg]);                   \
    }                                                                          \
  };                                                                           \
  asyncB(0, 0); wait_async0(); __syncthreads();                                \
  for (int kkI = 0; kkI < DIM/64; ++kkI) {                                     \
    const int kk = kkI*64;                                                     \
    const bf16_t* wbuf = &Wst[(kkI & 1)*4096];                                 \
    if (kkI < DIM/64 - 1) asyncB(kk + 64, (kkI & 1) ^ 1);                      \
    _Pragma("unroll") for (int f = 0; f < 2; ++f) {                            \
      v16bf af = make_frag(arow + kk + f*32 + g*8,                             \
                           arow + kk + f*32 + g*8 + 16);                       \
      v16bf bf[4];                                                             \
      _Pragma("unroll") for (int nb = 0; nb < 4; ++nb) {                       \
        const bf16_t* br = &wbuf[(nb*16 + lmod)*64 + f*32 + g*8];              \
        bf[nb] = make_frag(br, br + 16);                                       \
      }                                                                        \
      _Pragma("unroll") for (int nb = 0; nb < 4; ++nb)                         \
        acc[nb] = __builtin_amdgcn_wmma_f32_16x16x32_bf16(                     \
            false, af, false, bf[nb], (short)0, acc[nb], false, false);        \
    }                                                                          \
    if (kkI < DIM/64 - 1) { wait_async0(); __syncthreads(); }                  \
  }

// ---------------------------------------------------------------------------
// Kernel A: qkv = xb @ Wqt^T + b, routed to q/k [b,h,n,64] and vT [b,h,64,n].
// ---------------------------------------------------------------------------
__global__ __launch_bounds__(128)
void qkv_kernel(const bf16_t* __restrict__ xb, const bf16_t* __restrict__ Wqt,
                const float* __restrict__ bias,
                bf16_t* __restrict__ q, bf16_t* __restrict__ k,
                bf16_t* __restrict__ vT)
{
  __shared__ __align__(16) bf16_t Wst[2*64*64];      // double-buffered [n][k]
  const bf16_t* arow = xb + (size_t)(blockIdx.y*64 + (threadIdx.x >> 5)*16
                                     + (threadIdx.x & 15))*DIM;
  GEMM_CORE(Wqt)

  #pragma unroll
  for (int nb = 0; nb < 4; ++nb) {
    const int c  = nTile + nb*16 + lmod;
    const float bv = bias[c];
    const int t  = c >> 10, d = c & 1023;
    const int h  = d >> 6,  hd = d & 63;
    const float scl = (t == 0) ? 0.125f : 1.0f;      // fold 1/sqrt(HD) into q
    #pragma unroll
    for (int e = 0; e < 8; ++e) {
      int row  = mTile + wave*16 + e + 8*g;
      int b    = row >> 11, nseq = row & 2047;
      bf16_t val = (bf16_t)((acc[nb][e] + bv) * scl);
      size_t bhBase = (size_t)(b*NH + h);
      if (t == 2)                                     // V stored transposed
        vT[(bhBase*HD + hd)*SEQ + nseq] = val;
      else if (t == 1)
        k[(bhBase*SEQ + nseq)*HD + hd] = val;
      else
        q[(bhBase*SEQ + nseq)*HD + hd] = val;
    }
  }
}

// ---------------------------------------------------------------------------
// Kernel C: out = y @ Wpt^T + b_proj (fp32 output).
// ---------------------------------------------------------------------------
__global__ __launch_bounds__(128)
void proj_kernel(const bf16_t* __restrict__ y, const bf16_t* __restrict__ Wpt,
                 const float* __restrict__ bias, float* __restrict__ out)
{
  __shared__ __align__(16) bf16_t Wst[2*64*64];
  const bf16_t* arow = y + (size_t)(blockIdx.y*64 + (threadIdx.x >> 5)*16
                                    + (threadIdx.x & 15))*DIM;
  GEMM_CORE(Wpt)

  #pragma unroll
  for (int nb = 0; nb < 4; ++nb) {
    int c = nTile + nb*16 + lmod;
    float bv = bias[c];
    #pragma unroll
    for (int e = 0; e < 8; ++e) {
      int row = mTile + wave*16 + e + 8*g;
      out[(size_t)row*DIM + c] = acc[nb][e] + bv;
    }
  }
}

// ---------------------------------------------------------------------------
// Kernel B: flash attention per (bh, 64-row q tile). Causal; skips blocks
// above the diagonal; online softmax in f32; bf16 WMMA for QK^T and P@V.
// K/V double-buffered in LDS via ASYNC global->LDS copies under the compute.
// ---------------------------------------------------------------------------
__global__ __launch_bounds__(128)
void attn_kernel(const bf16_t* __restrict__ q, const bf16_t* __restrict__ k,
                 const bf16_t* __restrict__ vT, bf16_t* __restrict__ y)
{
  __shared__ __align__(16) bf16_t Klds[2*64*64];     // [key][hd] x2
  __shared__ __align__(16) bf16_t Vt[2*64*64];       // [hd][key] x2
  __shared__ __align__(16) bf16_t Plds[4*16*64];     // per-wave P [16][64]
  const int tid  = threadIdx.x;
  const int wave = tid >> 5, lane = tid & 31;
  const int lmod = lane & 15, g = lane >> 4;
  const int bh   = blockIdx.y;
  const int qTile = blockIdx.x;
  const size_t seqBase = (size_t)bh * SEQ;

  const bf16_t* qrow = q + (seqBase + qTile*64 + wave*16 + lmod)*HD;
  v16bf qa[2];
  #pragma unroll
  for (int f = 0; f < 2; ++f)
    qa[f] = make_frag(qrow + f*32 + g*8, qrow + f*32 + g*8 + 16);

  const unsigned kbase = lds_addr(Klds);
  const unsigned vbase = lds_addr(Vt);
  auto asyncKV = [&](int kb, int buf) {
    const bf16_t* ksrc = k + (seqBase + kb*64)*HD;
    #pragma unroll
    for (int i = 0; i < 4; ++i) {
      int e = tid + i*128;                            // 16B unit index
      async_ld16(kbase + (unsigned)(buf*4096 + e*8)*2u, ksrc + e*8);
    }
    const bf16_t* vsrc = vT + (size_t)bh*HD*SEQ + kb*64;
    #pragma unroll
    for (int i = 0; i < 4; ++i) {
      int e = tid + i*128; int hd = e >> 3, seg = (e & 7)*8;
      async_ld16(vbase + (unsigned)(buf*4096 + hd*64 + seg)*2u,
                 vsrc + (size_t)hd*SEQ + seg);
    }
  };

  float mr[8], lr[8];
  v8f o[4];
  #pragma unroll
  for (int e = 0; e < 8; ++e) { mr[e] = -3.0e38f; lr[e] = 0.0f; }
  #pragma unroll
  for (int nb = 0; nb < 4; ++nb) o[nb] = {};

  asyncKV(0, 0); wait_async0(); __syncthreads();

  for (int kb = 0; kb <= qTile; ++kb) {
    const bf16_t* kbuf = &Klds[(kb & 1)*4096];
    const bf16_t* vbuf = &Vt[(kb & 1)*4096];
    if (kb < qTile) asyncKV(kb + 1, (kb & 1) ^ 1);   // prefetch next block

    // S = Q @ K^T (contract over hd); batch fragment loads per K-half.
    v8f s[4];
    #pragma unroll
    for (int nb = 0; nb < 4; ++nb) s[nb] = {};
    #pragma unroll
    for (int f = 0; f < 2; ++f) {
      v16bf bf[4];
      #pragma unroll
      for (int nb = 0; nb < 4; ++nb) {
        const bf16_t* br = &kbuf[(nb*16 + lmod)*64 + f*32 + g*8];
        bf[nb] = make_frag(br, br + 16);
      }
      #pragma unroll
      for (int nb = 0; nb < 4; ++nb)
        s[nb] = __builtin_amdgcn_wmma_f32_16x16x32_bf16(
            false, qa[f], false, bf[nb], (short)0, s[nb], false, false);
    }

    if (kb == qTile) {                               // diagonal-block mask
      #pragma unroll
      for (int nb = 0; nb < 4; ++nb) {
        int kg = kb*64 + nb*16 + lmod;
        #pragma unroll
        for (int e = 0; e < 8; ++e) {
          int qg = qTile*64 + wave*16 + e + 8*g;
          if (kg > qg) s[nb][e] = -3.0e38f;
        }
      }
    }

    // Online softmax (row = e + 8*g lives across a 16-lane half).
    #pragma unroll
    for (int e = 0; e < 8; ++e) {
      float rm = fmaxf(fmaxf(s[0][e], s[1][e]), fmaxf(s[2][e], s[3][e]));
      rm = fmaxf(rm, __shfl_xor(rm, 1));
      rm = fmaxf(rm, __shfl_xor(rm, 2));
      rm = fmaxf(rm, __shfl_xor(rm, 4));
      rm = fmaxf(rm, __shfl_xor(rm, 8));
      float mn    = fmaxf(mr[e], rm);
      float alpha = __expf(mr[e] - mn);
      mr[e] = mn;
      float rs = 0.0f;
      #pragma unroll
      for (int nb = 0; nb < 4; ++nb) {
        float pe = __expf(s[nb][e] - mn);
        s[nb][e] = pe;                               // reuse score regs as P
        rs += pe;
      }
      rs += __shfl_xor(rs, 1);
      rs += __shfl_xor(rs, 2);
      rs += __shfl_xor(rs, 4);
      rs += __shfl_xor(rs, 8);
      lr[e] = lr[e]*alpha + rs;
      #pragma unroll
      for (int nb = 0; nb < 4; ++nb) o[nb][e] *= alpha;
    }

    // C-layout P -> A-layout via per-wave LDS (DS in-order within a wave).
    bf16_t* pw = &Plds[wave*1024];
    #pragma unroll
    for (int nb = 0; nb < 4; ++nb)
      #pragma unroll
      for (int e = 0; e < 8; ++e)
        pw[(e + 8*g)*64 + nb*16 + lmod] = (bf16_t)s[nb][e];

    // O += P @ V (contract over key); batched fragment loads.
    #pragma unroll
    for (int f = 0; f < 2; ++f) {
      const bf16_t* pr = &pw[lmod*64 + f*32 + g*8];
      v16bf pa = make_frag(pr, pr + 16);
      v16bf bf[4];
      #pragma unroll
      for (int nb = 0; nb < 4; ++nb) {
        const bf16_t* br = &vbuf[(nb*16 + lmod)*64 + f*32 + g*8];
        bf[nb] = make_frag(br, br + 16);
      }
      #pragma unroll
      for (int nb = 0; nb < 4; ++nb)
        o[nb] = __builtin_amdgcn_wmma_f32_16x16x32_bf16(
            false, pa, false, bf[nb], (short)0, o[nb], false, false);
    }

    if (kb < qTile) { wait_async0(); __syncthreads(); }
  }

  const int b = bh >> 4, h = bh & 15;
  #pragma unroll
  for (int e = 0; e < 8; ++e) {
    float inv = 1.0f / lr[e];
    int nseq = qTile*64 + wave*16 + e + 8*g;
    size_t rowo = ((size_t)b*SEQ + nseq)*DIM + h*HD;
    #pragma unroll
    for (int nb = 0; nb < 4; ++nb)
      y[rowo + nb*16 + lmod] = (bf16_t)(o[nb][e] * inv);
  }
}

// ---------------------------------------------------------------------------
extern "C" void kernel_launch(void* const* d_in, const int* in_sizes, int n_in,
                              void* d_out, int out_size, void* d_ws, size_t ws_size,
                              hipStream_t stream) {
  const float* x      = (const float*)d_in[0];
  const float* W_qkv  = (const float*)d_in[1];
  const float* b_qkv  = (const float*)d_in[2];
  const float* W_proj = (const float*)d_in[3];
  const float* b_proj = (const float*)d_in[4];
  float* out = (float*)d_out;

  char* ws = (char*)d_ws;
  const size_t MB8 = (size_t)NROWS*DIM*sizeof(bf16_t);       // 8 MB
  bf16_t* xb  = (bf16_t*)(ws);                                // 8 MB
  bf16_t* q   = (bf16_t*)(ws + MB8);                          // 8 MB
  bf16_t* k   = (bf16_t*)(ws + 2*MB8);                        // 8 MB
  bf16_t* vT  = (bf16_t*)(ws + 3*MB8);                        // 8 MB
  bf16_t* y   = (bf16_t*)(ws + 4*MB8);                        // 8 MB
  bf16_t* Wqt = (bf16_t*)(ws + 5*MB8);                        // 6 MB
  bf16_t* Wpt = (bf16_t*)(ws + 5*MB8 + (size_t)NQKV*DIM*sizeof(bf16_t)); // 2 MB

  cvt_x_kernel <<<(NROWS*DIM)/(8*256), 256, 0, stream>>>(x, xb);
  cvt_wt_kernel<<<dim3(NQKV/64, DIM/64), 256, 0, stream>>>(W_qkv, Wqt, DIM, NQKV);
  cvt_wt_kernel<<<dim3(DIM/64,  DIM/64), 256, 0, stream>>>(W_proj, Wpt, DIM, DIM);

  qkv_kernel <<<dim3(NQKV/64, NROWS/64), 128, 0, stream>>>(xb, Wqt, b_qkv, q, k, vT);
  attn_kernel<<<dim3(SEQ/64, BATCH*NH),  128, 0, stream>>>(q, k, vT, y);
  proj_kernel<<<dim3(DIM/64,  NROWS/64), 128, 0, stream>>>(y, Wpt, b_proj, out);
}
